// KoopmanPhiICNN_76596446757064
// MI455X (gfx1250) — compile-verified, matrix operand order checked by reading, not run on hardware
//
#include <hip/hip_runtime.h>
#include <hip/hip_bf16.h>
#include <cmath>

typedef __bf16 bf16_t;
typedef bf16_t v16bf __attribute__((ext_vector_type(16)));
typedef float  v8f   __attribute__((ext_vector_type(8)));
typedef unsigned int u32;
typedef u32 u32x4 __attribute__((ext_vector_type(4)));

struct Pack32 { u32x4 lo, hi; };   // 32 bytes == v16bf

#define BATCH 4096
#define XS 512
#define ZS 256
#define US 64
#define DD 832
#define HH 832
#define NHID 3
#define STEPS 64

#define BM 256
#define BN 64
#define BK 32
#define BKP 56   // padded LDS row stride (bf16 elems): 112B, 16B-aligned, conflict-free

#define TM 4     // 16-row tiles per wave (wave tile 64 x 32)
#define TN 2     // 16-col tiles per wave

__device__ __forceinline__ float softplusf(float x) {
  return (x > 15.f) ? x : log1pf(__expf(x));
}
__device__ __forceinline__ float sigmoidf_(float x) {
  return 1.f / (1.f + __expf(-x));
}

// One K-pass of C += A[M,K] * W[N,K]^T using v_wmma_f32_16x16x32_bf16.
// Block tile 256x64, 8 waves in a 4(M) x 2(N) grid, 64x32 per wave.
__device__ __forceinline__ void gemm_pair(
    const bf16_t* __restrict__ A, const bf16_t* __restrict__ W, int K,
    int m0, int n0, int tid, v8f acc[TM][TN],
    bf16_t* __restrict__ As, bf16_t* __restrict__ Ws)
{
  const int lane  = tid & 31;
  const int wid   = tid >> 5;
  const int waveM = wid >> 1;          // 0..3  (64 rows each)
  const int waveN = wid & 1;           // 0..1  (32 cols each)
  const int lrow  = lane & 15;
  const int khalf = lane >> 4;         // K-half select (contiguous k = 16*khalf + s, same order for A and W)

  const int arow = tid;                // 0..255 : one full 32-elem row per thread
  const int wrow = tid >> 2;           // 0..63
  const int wcol = (tid & 3) << 3;     // 0,8,16,24

  for (int k0 = 0; k0 < K; k0 += BK) {
    const u32x4* ag = reinterpret_cast<const u32x4*>(A + (size_t)(m0 + arow) * K + k0);
    u32x4 a0 = ag[0];
    u32x4 a1 = ag[1];
    u32x4 a2 = ag[2];
    u32x4 a3 = ag[3];
    const u32x4* wg = reinterpret_cast<const u32x4*>(W + (size_t)(n0 + wrow) * K + k0 + wcol);
    u32x4 w0 = wg[0];
    __syncthreads();
    *reinterpret_cast<u32x4*>(As + arow * BKP)      = a0;
    *reinterpret_cast<u32x4*>(As + arow * BKP + 8)  = a1;
    *reinterpret_cast<u32x4*>(As + arow * BKP + 16) = a2;
    *reinterpret_cast<u32x4*>(As + arow * BKP + 24) = a3;
    *reinterpret_cast<u32x4*>(Ws + wrow * BKP + wcol) = w0;
    __syncthreads();

    v16bf af[TM], wf[TN];
#pragma unroll
    for (int tm = 0; tm < TM; ++tm) {
      const u32x4* p = reinterpret_cast<const u32x4*>(
          As + (waveM * 64 + tm * 16 + lrow) * BKP + (khalf << 4));
      Pack32 t; t.lo = p[0]; t.hi = p[1];
      af[tm] = __builtin_bit_cast(v16bf, t);
    }
#pragma unroll
    for (int tn = 0; tn < TN; ++tn) {
      const u32x4* p = reinterpret_cast<const u32x4*>(
          Ws + (waveN * 32 + tn * 16 + lrow) * BKP + (khalf << 4));
      Pack32 t; t.lo = p[0]; t.hi = p[1];
      wf[tn] = __builtin_bit_cast(v16bf, t);
    }
#pragma unroll
    for (int tm = 0; tm < TM; ++tm)
#pragma unroll
      for (int tn = 0; tn < TN; ++tn)
        acc[tm][tn] = __builtin_amdgcn_wmma_f32_16x16x32_bf16(
            false, af[tm], false, wf[tn], (short)0, acc[tm][tn], false, false);
  }
}

// Epilogue modes:
//  0: outF = acc (+Cin)(+vecN[n])               (f32 store / accumulate)
//  1: outB = bf16(softplus(acc + Cin))          (forward hidden)
//  2: outB = bf16(vecN[n] * sigmoid(acc + Cin)) (forward last -> d3)
//  3: outB = bf16(acc * (1 - exp(-hsrc)))       (backward delta * sigma(a))
__global__ __launch_bounds__(256)
void gemm_wmma_kernel(const bf16_t* __restrict__ A1, const bf16_t* __restrict__ W1, int K1,
                      const bf16_t* __restrict__ A2, const bf16_t* __restrict__ W2, int K2,
                      const float* __restrict__ Cin, const float* __restrict__ vecN,
                      const bf16_t* __restrict__ hsrc,
                      float* __restrict__ outF, bf16_t* __restrict__ outB,
                      int N, int mode)
{
  __shared__ bf16_t As[BM * BKP];
  __shared__ bf16_t Ws[BN * BKP];
  const int tid = threadIdx.x;
  const int n0 = blockIdx.x * BN;
  const int m0 = blockIdx.y * BM;

  v8f acc[TM][TN] = {};
  gemm_pair(A1, W1, K1, m0, n0, tid, acc, As, Ws);
  if (K2 > 0) gemm_pair(A2, W2, K2, m0, n0, tid, acc, As, Ws);

  const int lane  = tid & 31;
  const int wid   = tid >> 5;
  const int waveM = wid >> 1;
  const int waveN = wid & 1;
  const int lrow  = lane & 15;
  const int lhi   = lane >> 4;

#pragma unroll
  for (int tm = 0; tm < TM; ++tm)
#pragma unroll
    for (int tn = 0; tn < TN; ++tn)
#pragma unroll
      for (int v = 0; v < 8; ++v) {
        const int m = m0 + waveM * 64 + tm * 16 + lhi * 8 + v;
        const int n = n0 + waveN * 32 + tn * 16 + lrow;
        const size_t idx = (size_t)m * N + n;
        float val = acc[tm][tn][v];
        if (Cin) val += Cin[idx];
        if (mode == 0) {
          if (vecN) val += vecN[n];
          outF[idx] = val;
        } else if (mode == 1) {
          outB[idx] = (bf16_t)softplusf(val);
        } else if (mode == 2) {
          outB[idx] = (bf16_t)(vecN[n] * sigmoidf_(val));
        } else {
          float h = (float)hsrc[idx];
          outB[idx] = (bf16_t)(val * (1.f - __expf(-h)));
        }
      }
}

// ---------------- elementwise helpers ----------------

__global__ void k_f32_to_bf(const float* __restrict__ in, bf16_t* __restrict__ out, long n) {
  long i = (long)blockIdx.x * blockDim.x + threadIdx.x;
  if (i < n) out[i] = (bf16_t)in[i];
}
__global__ void k_softplus_to_bf(const float* __restrict__ in, bf16_t* __restrict__ out, long n) {
  long i = (long)blockIdx.x * blockDim.x + threadIdx.x;
  if (i < n) out[i] = (bf16_t)softplusf(in[i]);
}
__global__ void k_softplus_f32(const float* __restrict__ in, float* __restrict__ out, long n) {
  long i = (long)blockIdx.x * blockDim.x + threadIdx.x;
  if (i < n) out[i] = softplusf(in[i]);
}
// out[r*KO + k] = bf16(in[r*CI + k]), k < KO  (column slice)
__global__ void k_slice_bf(const float* __restrict__ in, bf16_t* __restrict__ out,
                           int rows, int CI, int KO) {
  long i = (long)blockIdx.x * blockDim.x + threadIdx.x;
  if (i < (long)rows * KO) {
    int r = (int)(i / KO), k = (int)(i % KO);
    out[i] = (bf16_t)in[(size_t)r * CI + k];
  }
}
// out[n*KO + k] = cvt(in[k*CI + n]) (transpose, optional softplus)
__global__ void k_transpose_bf(const float* __restrict__ in, bf16_t* __restrict__ out,
                               int NO, int KO, int CI, int do_sp) {
  long i = (long)blockIdx.x * blockDim.x + threadIdx.x;
  if (i < (long)NO * KO) {
    int n = (int)(i / KO), k = (int)(i % KO);
    float v = in[(size_t)k * CI + n];
    if (do_sp) v = softplusf(v);
    out[i] = (bf16_t)v;
  }
}
__global__ void k_build_y0(const float* __restrict__ x, const float* __restrict__ u,
                           bf16_t* __restrict__ y0, long n) {
  long i = (long)blockIdx.x * blockDim.x + threadIdx.x;
  if (i < n) {
    int b = (int)(i / DD), d = (int)(i % DD);
    float v = 0.f;
    if (d >= ZS && d < ZS + XS) v = x[(size_t)b * XS + (d - ZS)];
    else if (d >= ZS + XS)      v = u[(size_t)b * US + (d - ZS - XS)];
    y0[i] = (bf16_t)v;
  }
}
__global__ void k_zero_f32(float* __restrict__ p, long n) {
  long i = (long)blockIdx.x * blockDim.x + threadIdx.x;
  if (i < n) p[i] = 0.f;
}
__global__ void k_zero_bf(bf16_t* __restrict__ p, long n) {
  long i = (long)blockIdx.x * blockDim.x + threadIdx.x;
  if (i < n) p[i] = (bf16_t)0.f;
}
__global__ void k_adam(const float* __restrict__ gz, const float* __restrict__ wx_out,
                       float* __restrict__ z, float* __restrict__ m, float* __restrict__ v,
                       bf16_t* __restrict__ zbf, float c1inv, float c2inv, long n) {
  long i = (long)blockIdx.x * blockDim.x + threadIdx.x;
  if (i < n) {
    int col = (int)(i & (ZS - 1));
    float g = gz[i] + wx_out[col];
    float mi = 0.9f * m[i] + 0.1f * g;
    float vi = 0.999f * v[i] + 0.001f * g * g;
    m[i] = mi; v[i] = vi;
    float mh = mi * c1inv;
    float vh = vi * c2inv;
    float zi = z[i] - 0.01f * mh / (sqrtf(vh) + 1e-8f);
    z[i] = zi;
    zbf[i] = (bf16_t)zi;
  }
}

// ---------------- host side ----------------

extern "C" void kernel_launch(void* const* d_in, const int* in_sizes, int n_in,
                              void* d_out, int out_size, void* d_ws, size_t ws_size,
                              hipStream_t stream) {
  (void)in_sizes; (void)n_in; (void)out_size; (void)ws_size;
  const float* x      = (const float*)d_in[0];
  const float* u      = (const float*)d_in[1];
  const float* W_in   = (const float*)d_in[2];
  const float* b_in   = (const float*)d_in[3];
  const float* Wzs    = (const float*)d_in[4];
  const float* Wxs    = (const float*)d_in[5];
  const float* bs     = (const float*)d_in[6];
  const float* wz_out = (const float*)d_in[7];
  const float* wx_out = (const float*)d_in[8];

  char* p = (char*)d_ws;
  auto take = [&](size_t bytes) -> char* {
    char* r = p; p += (bytes + 255) & ~(size_t)255; return r;
  };
  const size_t BH = (size_t)BATCH * HH, BD = (size_t)BATCH * DD, BZ = (size_t)BATCH * ZS;
  bf16_t* y0_bf    = (bf16_t*)take(BD * 2);
  bf16_t* W_in_bf  = (bf16_t*)take((size_t)HH * DD * 2);
  bf16_t* Wx_bf    = (bf16_t*)take((size_t)NHID * HH * DD * 2);
  bf16_t* spWz_bf  = (bf16_t*)take((size_t)NHID * HH * HH * 2);
  bf16_t* spWzT_bf = (bf16_t*)take((size_t)NHID * HH * HH * 2);
  bf16_t* Winz_bf  = (bf16_t*)take((size_t)HH * ZS * 2);
  bf16_t* Wxz_bf   = (bf16_t*)take((size_t)NHID * HH * ZS * 2);
  bf16_t* WinzT_bf = (bf16_t*)take((size_t)ZS * HH * 2);
  bf16_t* WxzT_bf  = (bf16_t*)take((size_t)NHID * ZS * HH * 2);
  float*  spwz_f   = (float*)take((size_t)HH * 4);
  float*  c_in     = (float*)take(BH * 4);
  float*  cC       = (float*)take((size_t)NHID * BH * 4);
  bf16_t* hbuf     = (bf16_t*)take((size_t)NHID * BH * 2);
  bf16_t* dA       = (bf16_t*)take(BH * 2);
  bf16_t* dB       = (bf16_t*)take(BH * 2);
  float*  zf       = (float*)take(3 * BZ * 4);  // z, m, v contiguous
  float*  mf       = zf + BZ;
  float*  vf       = mf + BZ;
  bf16_t* zbf      = (bf16_t*)take(BZ * 2);
  float*  gz       = (float*)take(BZ * 4);

  auto ew = [&](long n) { return dim3((unsigned)((n + 255) / 256)); };
  auto gemm = [&](const bf16_t* A1, const bf16_t* W1, int K1,
                  const bf16_t* A2, const bf16_t* W2, int K2,
                  const float* Cin, const float* vecN, const bf16_t* hsrc,
                  float* outF, bf16_t* outB, int N, int mode) {
    dim3 grid(N / BN, BATCH / BM);
    gemm_wmma_kernel<<<grid, 256, 0, stream>>>(A1, W1, K1, A2, W2, K2,
                                               Cin, vecN, hsrc, outF, outB, N, mode);
  };

  // ---- one-time prep ----
  k_build_y0<<<ew(BD), 256, 0, stream>>>(x, u, y0_bf, (long)BD);
  k_f32_to_bf<<<ew((long)HH * DD), 256, 0, stream>>>(W_in, W_in_bf, (long)HH * DD);
  k_f32_to_bf<<<ew((long)NHID * HH * DD), 256, 0, stream>>>(Wxs, Wx_bf, (long)NHID * HH * DD);
  k_softplus_to_bf<<<ew((long)NHID * HH * HH), 256, 0, stream>>>(Wzs, spWz_bf, (long)NHID * HH * HH);
  for (int l = 0; l < NHID; ++l)
    k_transpose_bf<<<ew((long)HH * HH), 256, 0, stream>>>(
        Wzs + (size_t)l * HH * HH, spWzT_bf + (size_t)l * HH * HH, HH, HH, HH, 1);
  k_slice_bf<<<ew((long)HH * ZS), 256, 0, stream>>>(W_in, Winz_bf, HH, DD, ZS);
  k_slice_bf<<<ew((long)NHID * HH * ZS), 256, 0, stream>>>(Wxs, Wxz_bf, NHID * HH, DD, ZS);
  k_transpose_bf<<<ew((long)ZS * HH), 256, 0, stream>>>(W_in, WinzT_bf, ZS, HH, DD, 0);
  for (int l = 0; l < NHID; ++l)
    k_transpose_bf<<<ew((long)ZS * HH), 256, 0, stream>>>(
        Wxs + (size_t)l * HH * DD, WxzT_bf + (size_t)l * ZS * HH, ZS, HH, DD, 0);
  k_softplus_f32<<<ew(HH), 256, 0, stream>>>(wz_out, spwz_f, HH);
  k_zero_f32<<<ew(3 * (long)BZ), 256, 0, stream>>>(zf, 3 * (long)BZ);
  k_zero_bf<<<ew((long)BZ), 256, 0, stream>>>(zbf, (long)BZ);

  // ---- precompute fixed x/u contributions (f32) ----
  gemm(y0_bf, W_in_bf, DD, nullptr, nullptr, 0, nullptr, b_in, nullptr, c_in, nullptr, HH, 0);
  for (int l = 0; l < NHID; ++l)
    gemm(y0_bf, Wx_bf + (size_t)l * HH * DD, DD, nullptr, nullptr, 0,
         nullptr, bs + (size_t)l * HH, nullptr, cC + (size_t)l * BH, nullptr, HH, 0);

  // ---- 64 Adam steps ----
  bf16_t* h0 = hbuf;
  bf16_t* h1 = hbuf + BH;
  bf16_t* h2 = hbuf + 2 * BH;
  for (int t = 0; t < STEPS; ++t) {
    // forward
    gemm(zbf, Winz_bf, ZS, nullptr, nullptr, 0, c_in, nullptr, nullptr, nullptr, h0, HH, 1);
    gemm(h0, spWz_bf, HH, zbf, Wxz_bf, ZS, cC, nullptr, nullptr, nullptr, h1, HH, 1);
    gemm(h1, spWz_bf + (size_t)HH * HH, HH, zbf, Wxz_bf + (size_t)HH * ZS, ZS,
         cC + BH, nullptr, nullptr, nullptr, h2, HH, 1);
    gemm(h2, spWz_bf + 2 * (size_t)HH * HH, HH, zbf, Wxz_bf + 2 * (size_t)HH * ZS, ZS,
         cC + 2 * BH, spwz_f, nullptr, nullptr, dA, HH, 2);   // dA = d3
    // backward
    gemm(dA, WxzT_bf + 2 * (size_t)ZS * HH, HH, nullptr, nullptr, 0,
         nullptr, nullptr, nullptr, gz, nullptr, ZS, 0);                    // gz  = d3 @ Wxz2
    gemm(dA, spWzT_bf + 2 * (size_t)HH * HH, HH, nullptr, nullptr, 0,
         nullptr, nullptr, h2, nullptr, dB, HH, 3);                         // dB  = d2
    gemm(dB, WxzT_bf + (size_t)ZS * HH, HH, nullptr, nullptr, 0,
         gz, nullptr, nullptr, gz, nullptr, ZS, 0);                         // gz += d2 @ Wxz1
    gemm(dB, spWzT_bf + (size_t)HH * HH, HH, nullptr, nullptr, 0,
         nullptr, nullptr, h1, nullptr, dA, HH, 3);                         // dA  = d1
    gemm(dA, WxzT_bf, HH, nullptr, nullptr, 0,
         gz, nullptr, nullptr, gz, nullptr, ZS, 0);                         // gz += d1 @ Wxz0
    gemm(dA, spWzT_bf, HH, nullptr, nullptr, 0,
         nullptr, nullptr, h0, nullptr, dB, HH, 3);                         // dB  = d0
    gemm(dB, WinzT_bf, HH, nullptr, nullptr, 0,
         gz, nullptr, nullptr, gz, nullptr, ZS, 0);                         // gz += d0 @ Winz
    // Adam update
    float c1inv = (float)(1.0 / (1.0 - std::pow(0.9,   (double)(t + 1))));
    float c2inv = (float)(1.0 / (1.0 - std::pow(0.999, (double)(t + 1))));
    k_adam<<<ew((long)BZ), 256, 0, stream>>>(gz, wx_out, zf, mf, vf, zbf, c1inv, c2inv, (long)BZ);
  }

  hipMemcpyAsync(d_out, zf, BZ * sizeof(float), hipMemcpyDeviceToDevice, stream);
}
